// Model_54571854463036
// MI455X (gfx1250) — matrix operand order repaired and checked by reading, not compiled, not run on hardware
//
#include <hip/hip_runtime.h>
#include <hip/hip_bf16.h>
#include <math.h>

// ---------------------------------------------------------------------------
// CDNA5 (gfx1250) wave32 WMMA building blocks.
// Fragment layouts used by these helpers (contiguous k, not the ISA 7.12.2 table order):
//  A (16x32 f16): lane m = l&15, half-base kb = (l>>4)*16;
//                 element i -> K = kb+i
//  B (32x16 f16): lane n = l&15, kb = (l>>4)*16; element i -> K = kb+i (same order as A)
//  C/D (16x16 f32): vgpr r -> row M = r + (l<16 ? 0 : 8), col N = l&15
// ---------------------------------------------------------------------------
typedef __attribute__((ext_vector_type(16))) _Float16 v16h;
typedef __attribute__((ext_vector_type(2)))  _Float16 v2h;
typedef __attribute__((ext_vector_type(8)))  float    v8f;

__device__ __forceinline__ v8f wmma16(v16h a, v16h b, v8f c) {
  return __builtin_amdgcn_wmma_f32_16x16x32_f16(false, a, false, b, (short)0, c,
                                                false, false);
}

// A fragment from f32 row-major source; K-pairs are contiguous -> b64 loads.
__device__ __forceinline__ v16h load_a_frag_g(const float* __restrict__ src, int ld) {
  const int lane = threadIdx.x & 31;
  const float* row = src + (size_t)(lane & 15) * ld + ((lane >> 4) * 16);
  v16h a;
#pragma unroll
  for (int j = 0; j < 8; ++j) {
    const int kb = 2 * j;
    const float2 p = *(const float2*)(row + kb);
    a[2 * j]     = (_Float16)p.x;
    a[2 * j + 1] = (_Float16)p.y;
  }
  return a;
}

// A fragment from f16 LDS tile (row stride 32 f16), pairs contiguous.
__device__ __forceinline__ v16h load_a_frag_lds(const _Float16* src) {
  const int lane = threadIdx.x & 31;
  const _Float16* row = src + (lane & 15) * 32 + ((lane >> 4) * 16);
  v16h a;
#pragma unroll
  for (int j = 0; j < 8; ++j) {
    const int kb = 2 * j;
    const v2h p = *(const v2h*)(row + kb);
    a[2 * j]     = p.x;
    a[2 * j + 1] = p.y;
  }
  return a;
}

// B fragment, strided source; logical B[k][n] = src[k*ld + n] (used for V).
__device__ __forceinline__ v16h load_b_frag_g(const float* __restrict__ src, int ld) {
  const int lane = threadIdx.x & 31;
  const float* col = src + (lane & 15) + (size_t)((lane >> 4) * 16) * ld;
  v16h b;
#pragma unroll
  for (int i = 0; i < 16; ++i) b[i] = (_Float16)col[(size_t)i * ld];
  return b;
}

// B fragment from transposed source (K^T): 16 contiguous floats -> b128 loads.
__device__ __forceinline__ v16h load_bt_frag_g(const float* __restrict__ src, int ld) {
  const int lane = threadIdx.x & 31;
  const float* row = src + (size_t)(lane & 15) * ld + ((lane >> 4) * 16);
  v16h b;
#pragma unroll
  for (int q = 0; q < 4; ++q) {
    const float4 p = *(const float4*)(row + 4 * q);
    b[4 * q]     = (_Float16)p.x;
    b[4 * q + 1] = (_Float16)p.y;
    b[4 * q + 2] = (_Float16)p.z;
    b[4 * q + 3] = (_Float16)p.w;
  }
  return b;
}

// B fragment from pre-swizzled LDS: lane's 16 f16 are 32 contiguous bytes.
__device__ __forceinline__ v16h load_b_frag_lds(const _Float16* p) {
  v16h b;
#pragma unroll
  for (int i = 0; i < 16; ++i) b[i] = p[i];
  return b;
}

// ---------------------------------------------------------------------------
// Batched GEMM, Nout fixed at 128: Y[b] = epilogue(X[b] @ W + bias).
// Block = WPB waves; each wave owns a 16x128 row-slab (8 WMMA accumulators,
// A fragment reused 8x). The 32x128 W k-slab is cooperatively staged to LDS
// once per block, converted to f16 and stored pre-swizzled in B-fragment
// order so each lane's fragment is two ds_load_b128.
// EPI 0: Y = gemm+b ; EPI 1: Y = X + relu(gemm+b) ; EPI 2: relu(X + relu(gemm+b))
// (residual epilogues require Kd == 128)
// ---------------------------------------------------------------------------
template <int EPI>
__global__ __launch_bounds__(256) void gemm_bias_kernel(
    float* __restrict__ Y, const float* __restrict__ X, const float* __restrict__ W,
    const float* __restrict__ bias, int M, int Kd, int WPB) {
  constexpr int NOUT = 128;
  __shared__ _Float16 wtile[32 * NOUT];  // [tile t][lane l][i] = (t*32+l)*16+i

  const int wave = threadIdx.x >> 5;
  const int lane = threadIdx.x & 31;
  const int b = blockIdx.y;
  const int mbase = (blockIdx.x * WPB + wave) * 16;
  const float* Xb = X + (size_t)b * M * Kd;
  float* Yb = Y + (size_t)b * M * NOUT;
  const int ln = lane & 15;
  const int off8 = (lane < 16) ? 0 : 8;

  v8f acc[8];
  const v8f vzero = {};
#pragma unroll
  for (int t = 0; t < 8; ++t) acc[t] = vzero;

  for (int k0 = 0; k0 < Kd; k0 += 32) {
    // cooperative, coalesced W slab load -> f16, swizzled to B-fragment order
    for (int e = threadIdx.x; e < 32 * NOUT; e += blockDim.x) {
      const int kk = e >> 7, nn = e & 127;
      const int t = nn >> 4, n = nn & 15, half = kk >> 4, i = kk & 15;
      wtile[((t * 32) + half * 16 + n) * 16 + i] =
          (_Float16)W[(size_t)(k0 + kk) * NOUT + nn];
    }
    __syncthreads();
    __builtin_prefetch(Xb + (size_t)mbase * Kd + k0 + 32, 0, 0);
    const v16h a = load_a_frag_g(Xb + (size_t)mbase * Kd + k0, Kd);
#pragma unroll
    for (int t = 0; t < 8; ++t)
      acc[t] = wmma16(a, load_b_frag_lds(wtile + (t * 32 + lane) * 16), acc[t]);
    __syncthreads();
  }

#pragma unroll
  for (int t = 0; t < 8; ++t) {
    const int n0 = t * 16;
    const float bn = bias[n0 + ln];
#pragma unroll
    for (int r = 0; r < 8; ++r) {
      const int m = mbase + r + off8;
      float y = acc[t][r] + bn;
      if (EPI == 1) y = Xb[(size_t)m * Kd + n0 + ln] + fmaxf(y, 0.f);
      else if (EPI == 2) y = fmaxf(Xb[(size_t)m * Kd + n0 + ln] + fmaxf(y, 0.f), 0.f);
      Yb[(size_t)m * NOUT + n0 + ln] = y;
    }
  }
}

// ---------------------------------------------------------------------------
// Flash attention per (batch, head, 16-query tile); one wave per block.
// DH: head dim (32 for MABs, 128 for IAA).  MMODE: 0 none,
//   1 idx-mask (allowed iff idx[b,key]==query_row && msk[b,key]!=0),
//   2 pair-mask (allowed iff msk[b,q]!=0 && msk[b,key]!=0).
// RESIDQ: add q residual (MAB: o = q + softmax(s)@v).
// Replicates reference: s = s*scale; masked -> -1e10; softmax over keys.
// ---------------------------------------------------------------------------
template <int DH, int MMODE, bool RESIDQ>
__global__ __launch_bounds__(32) void attn_kernel(
    float* __restrict__ O, const float* __restrict__ Q, const float* __restrict__ K,
    const float* __restrict__ V, const int* __restrict__ idx,
    const float* __restrict__ msk, int NQ, int NK, float scale) {
  constexpr int DM = 128;        // model dim (row stride of q/k/v buffers)
  constexpr int DCH = DH / 32;   // K-chunks for q@k^T
  constexpr int ONT = DH / 16;   // 16-wide output column tiles
  const int q0 = blockIdx.x * 16, h = blockIdx.y, b = blockIdx.z;
  const int lane = threadIdx.x;
  const int lm = lane & 15;
  const int off8 = (lane < 16) ? 0 : 8;

  const float* Qb = Q + ((size_t)b * NQ) * DM + h * DH;
  const float* Kb = K + ((size_t)b * NK) * DM + h * DH;
  const float* Vb = V + ((size_t)b * NK) * DM + h * DH;

  v16h aq[DCH];
#pragma unroll
  for (int c = 0; c < DCH; ++c)
    aq[c] = load_a_frag_g(Qb + (size_t)q0 * DM + c * 32, DM);

  __shared__ _Float16 pbuf[16 * 32];

  float rmax[8], rsum[8], mq[8];
  v8f oacc[ONT];
  const v8f vzero = {};
#pragma unroll
  for (int r = 0; r < 8; ++r) { rmax[r] = -INFINITY; rsum[r] = 0.f; mq[r] = 1.f; }
#pragma unroll
  for (int j = 0; j < ONT; ++j) oacc[j] = vzero;
  if (MMODE == 2) {
#pragma unroll
    for (int r = 0; r < 8; ++r) mq[r] = msk[(size_t)b * NK + q0 + r + off8];
  }

  for (int k0 = 0; k0 < NK; k0 += 32) {
    v8f s[2];
#pragma unroll
    for (int half = 0; half < 2; ++half) {
      v8f sa = vzero;
#pragma unroll
      for (int c = 0; c < DCH; ++c)
        sa = wmma16(aq[c],
                    load_bt_frag_g(Kb + (size_t)(k0 + half * 16) * DM + c * 32, DM), sa);
      s[half] = sa;
    }
    // scale + mask (element (m=r+off8, key=k0+half*16+lm))
#pragma unroll
    for (int half = 0; half < 2; ++half) {
      const int key = k0 + half * 16 + lm;
      float mk = 1.f; int ki = -1;
      if (MMODE == 1) { ki = idx[(size_t)b * NK + key]; mk = msk[(size_t)b * NK + key]; }
      if (MMODE == 2) { mk = msk[(size_t)b * NK + key]; }
#pragma unroll
      for (int r = 0; r < 8; ++r) {
        float sv = s[half][r] * scale;
        if (MMODE == 1 && !(ki == (q0 + r + off8) && mk != 0.f)) sv = -1e10f;
        if (MMODE == 2 && !(mk != 0.f && mq[r] != 0.f)) sv = -1e10f;
        s[half][r] = sv;
      }
    }
    // online softmax; row stats replicated across each 16-lane half
#pragma unroll
    for (int r = 0; r < 8; ++r) {
      float vm = fmaxf(s[0][r], s[1][r]);
#pragma unroll
      for (int d = 1; d < 16; d <<= 1) vm = fmaxf(vm, __shfl_xor(vm, d, 32));
      const float mnew = fmaxf(rmax[r], vm);
      const float alpha = __expf(rmax[r] - mnew);
      rmax[r] = mnew;
      const float p0 = __expf(s[0][r] - mnew);
      const float p1 = __expf(s[1][r] - mnew);
      float ps = p0 + p1;
#pragma unroll
      for (int d = 1; d < 16; d <<= 1) ps += __shfl_xor(ps, d, 32);
      rsum[r] = rsum[r] * alpha + ps;
#pragma unroll
      for (int j = 0; j < ONT; ++j) oacc[j][r] *= alpha;
      pbuf[(r + off8) * 32 + lm]      = (_Float16)p0;
      pbuf[(r + off8) * 32 + 16 + lm] = (_Float16)p1;
    }
    __syncthreads();
    const v16h pf = load_a_frag_lds(pbuf);   // C-layout -> A-layout via LDS
#pragma unroll
    for (int j = 0; j < ONT; ++j)
      oacc[j] = wmma16(pf, load_b_frag_g(Vb + (size_t)k0 * DM + j * 16, DM), oacc[j]);
    __syncthreads();
  }
  // normalize, optional q residual, store
#pragma unroll
  for (int j = 0; j < ONT; ++j) {
#pragma unroll
    for (int r = 0; r < 8; ++r) {
      const int m = q0 + r + off8;
      float o = oacc[j][r] / rsum[r];
      if (RESIDQ) o += Qb[(size_t)m * DM + j * 16 + lm];
      O[((size_t)b * NQ + m) * DM + h * DH + j * 16 + lm] = o;
    }
  }
}

// ---------------------------------------------------------------------------
// Gather/copy rows into strided destination (concatenated K inputs).
// dst[b,n, dstOff : dstOff+D] = src[b, idx ? idx[b,n] : n, :]
// ---------------------------------------------------------------------------
__global__ void gather_rows_kernel(float* __restrict__ dst, int dstStride, int dstOff,
                                   const float* __restrict__ src,
                                   const int* __restrict__ idx, int Bn, int Nn, int Dd,
                                   int srcRows) {
  const long long t = (long long)blockIdx.x * blockDim.x + threadIdx.x;
  const long long total = (long long)Bn * Nn * Dd;
  if (t >= total) return;
  const int d = (int)(t % Dd);
  const long long rd = t / Dd;
  const int b = (int)(rd / Nn);
  const int n = (int)(rd % Nn);
  const int srow = idx ? idx[rd] : n;
  dst[rd * (long long)dstStride + dstOff + d] =
      src[((size_t)b * srcRows + srow) * Dd + d];
}

// ---------------------------------------------------------------------------
// Encoder kernels
// ---------------------------------------------------------------------------
__global__ void encode_obs_kernel(float* __restrict__ obs, const float* __restrict__ x,
                                  const float* __restrict__ msk,
                                  const float* __restrict__ ymsk,
                                  const float* __restrict__ w,
                                  const float* __restrict__ bv, long long BN, int Dd) {
  const long long t = (long long)blockIdx.x * blockDim.x + threadIdx.x;
  if (t >= BN * Dd) return;
  const long long r = t / Dd;
  const int d = (int)(t % Dd);
  const float m = msk[r];
  const float a0 = x[r];
  const float a1 = 1.f - m + ymsk[r];
  const float v = a0 * w[d] + a1 * w[Dd + d] + bv[d];
  obs[t] = fmaxf(v, 0.f) * m;
}

__global__ void encode_temp_kernel(float* __restrict__ temp, const float* __restrict__ mark,
                                   const float* __restrict__ w,
                                   const float* __restrict__ bv, long long BL, int Dd) {
  const long long t = (long long)blockIdx.x * blockDim.x + threadIdx.x;
  if (t >= BL * Dd) return;
  const long long r = t / Dd;
  const int d = (int)(t % Dd);
  temp[t] = __sinf(mark[r] * w[d] + bv[d]);
}

__global__ void encode_var_kernel(float* __restrict__ var, const float* __restrict__ w,
                                  long long total, int EDd) {
  const long long t = (long long)blockIdx.x * blockDim.x + threadIdx.x;
  if (t >= total) return;
  var[t] = fmaxf(w[t % EDd], 0.f);
}

// ---------------------------------------------------------------------------
// Host orchestration
// ---------------------------------------------------------------------------
static inline int cdiv(long long a, int b) { return (int)((a + b - 1) / b); }

extern "C" void kernel_launch(void* const* d_in, const int* in_sizes, int n_in,
                              void* d_out, int out_size, void* d_ws, size_t ws_size,
                              hipStream_t stream) {
  (void)in_sizes; (void)n_in; (void)out_size; (void)ws_size;
  constexpr int B = 16, N = 1024, D = 128, L = 64, E = 32, NL = 2, H = 4;
  const float scale = 1.0f / sqrtf((float)D);

  const float* x     = (const float*)d_in[0];
  const float* maskp = (const float*)d_in[1];
  const float* ymask = (const float*)d_in[2];
  const float* mark  = (const float*)d_in[3];
  const float* w_obs = (const float*)d_in[4];
  const float* b_obs = (const float*)d_in[5];
  const float* w_tm  = (const float*)d_in[6];
  const float* b_tm  = (const float*)d_in[7];
  const float* var_w = (const float*)d_in[8];
  const float *tq_w=(const float*)d_in[9],  *tq_b=(const float*)d_in[10];
  const float *tk_w=(const float*)d_in[11], *tk_b=(const float*)d_in[12];
  const float *tv_w=(const float*)d_in[13], *tv_b=(const float*)d_in[14];
  const float *to_w=(const float*)d_in[15], *to_b=(const float*)d_in[16];
  const float *vq_w=(const float*)d_in[17], *vq_b=(const float*)d_in[18];
  const float *vk_w=(const float*)d_in[19], *vk_b=(const float*)d_in[20];
  const float *vv_w=(const float*)d_in[21], *vv_b=(const float*)d_in[22];
  const float *vo_w=(const float*)d_in[23], *vo_b=(const float*)d_in[24];
  const float *sq_w=(const float*)d_in[25], *sq_b=(const float*)d_in[26];
  const float *sk_w=(const float*)d_in[27], *sk_b=(const float*)d_in[28];
  const float *sv_w=(const float*)d_in[29], *sv_b=(const float*)d_in[30];
  const float *so_w=(const float*)d_in[31], *so_b=(const float*)d_in[32];
  const float *iq_w=(const float*)d_in[33], *iq_b=(const float*)d_in[34];
  const float *ik_w=(const float*)d_in[35], *ik_b=(const float*)d_in[36];
  const float *iv_w=(const float*)d_in[37], *iv_b=(const float*)d_in[38];
  const int* vidx = (const int*)d_in[39];
  const int* tidx = (const int*)d_in[40];

  const size_t oBND = (size_t)B * N * D;
  const size_t oBLD = (size_t)B * L * D;
  const size_t oBED = (size_t)B * E * D;

  float* ws = (float*)d_ws;
  float* obsA  = ws;
  float* obsB  = obsA + oBND;
  float* tempA = obsB + oBND;
  float* tempB = tempA + oBLD;
  float* varA  = tempB + oBLD;
  float* varB  = varA + oBED;
  float* kcat  = varB + oBED;            // B*N*3D
  float* qbuf  = kcat + (size_t)B * N * 3 * D;
  float* kbuf  = qbuf + oBND;
  float* vbuf  = kbuf + oBND;
  float* obuf  = kcat;                   // alias: kcat dead once attn runs

  auto gemm = [&](int epi, float* Y, const float* Xp, const float* Wp,
                  const float* bias, int M, int Kd) {
    int WPB = M / 16; if (WPB > 8) WPB = 8;
    dim3 g(M / (16 * WPB), B);
    dim3 blk(32 * WPB);
    if (epi == 0)
      gemm_bias_kernel<0><<<g, blk, 0, stream>>>(Y, Xp, Wp, bias, M, Kd, WPB);
    else if (epi == 1)
      gemm_bias_kernel<1><<<g, blk, 0, stream>>>(Y, Xp, Wp, bias, M, Kd, WPB);
    else
      gemm_bias_kernel<2><<<g, blk, 0, stream>>>(Y, Xp, Wp, bias, M, Kd, WPB);
  };
  auto gather = [&](float* dst, int stride, int offc, const float* src, const int* idx,
                    int srcRows) {
    const long long tot = (long long)B * N * D;
    gather_rows_kernel<<<cdiv(tot, 256), 256, 0, stream>>>(dst, stride, offc, src, idx,
                                                           B, N, D, srcRows);
  };

  // ---- HypergraphEncoder ----
  encode_obs_kernel<<<cdiv((long long)B * N * D, 256), 256, 0, stream>>>(
      obsA, x, maskp, ymask, w_obs, b_obs, (long long)B * N, D);
  encode_temp_kernel<<<cdiv((long long)B * L * D, 256), 256, 0, stream>>>(
      tempA, mark, w_tm, b_tm, (long long)B * L, D);
  encode_var_kernel<<<cdiv((long long)B * E * D, 256), 256, 0, stream>>>(
      varA, var_w, (long long)B * E * D, E * D);

  float *obsCur = obsA, *obsNew = obsB;
  float *tempCur = tempA, *tempNew = tempB;
  float *varCur = varA, *varNew = varB;

  for (int i = 0; i < NL; ++i) {
    const size_t wDD = (size_t)i * D * D, w2DD = (size_t)i * 2 * D * D,
                 w3DD = (size_t)i * 3 * D * D, wD = (size_t)i * D;
    // ---- temporal MAB: K = [var_g, obs] ----
    gather(kcat, 2 * D, 0, varCur, vidx, E);
    gather(kcat, 2 * D, D, obsCur, nullptr, N);
    gemm(0, qbuf, tempCur, tq_w + wDD, tq_b + wD, L, D);
    gemm(0, kbuf, kcat, tk_w + w2DD, tk_b + wD, N, 2 * D);
    gemm(0, vbuf, kcat, tv_w + w2DD, tv_b + wD, N, 2 * D);
    attn_kernel<32, 1, true><<<dim3(L / 16, H, B), 32, 0, stream>>>(
        obuf, qbuf, kbuf, vbuf, tidx, maskp, L, N, scale);
    gemm(1, tempNew, obuf, to_w + wDD, to_b + wD, L, D);
    // ---- variable MAB: K = [temp_g(old), obs] ----
    gather(kcat, 2 * D, 0, tempCur, tidx, L);
    gather(kcat, 2 * D, D, obsCur, nullptr, N);
    gemm(0, qbuf, varCur, vq_w + wDD, vq_b + wD, E, D);
    gemm(0, kbuf, kcat, vk_w + w2DD, vk_b + wD, N, 2 * D);
    gemm(0, vbuf, kcat, vv_w + w2DD, vv_b + wD, N, 2 * D);
    attn_kernel<32, 1, true><<<dim3(E / 16, H, B), 32, 0, stream>>>(
        obuf, qbuf, kbuf, vbuf, vidx, maskp, E, N, scale);
    gemm(1, varNew, obuf, vo_w + wDD, vo_b + wD, E, D);
    { float* t = tempCur; tempCur = tempNew; tempNew = t; }
    { float* t = varCur;  varCur  = varNew;  varNew  = t; }
    // ---- obs self MAB: K = [temp_g(new), var_g(new), obs] ----
    gather(kcat, 3 * D, 0, tempCur, tidx, L);
    gather(kcat, 3 * D, D, varCur, vidx, E);
    gather(kcat, 3 * D, 2 * D, obsCur, nullptr, N);
    gemm(0, qbuf, obsCur, sq_w + wDD, sq_b + wD, N, D);
    gemm(0, kbuf, kcat, sk_w + w3DD, sk_b + wD, N, 3 * D);
    gemm(0, vbuf, kcat, sv_w + w3DD, sv_b + wD, N, 3 * D);
    attn_kernel<32, 2, true><<<dim3(N / 16, H, B), 32, 0, stream>>>(
        obuf, qbuf, kbuf, vbuf, nullptr, maskp, N, N, scale);
    gemm(2, obsNew, obuf, so_w + wDD, so_b + wD, N, D);
    { float* t = obsCur; obsCur = obsNew; obsNew = t; }
    // ---- IAA on var hyperedges at last layer ----
    if (i == NL - 1) {
      gemm(0, qbuf, varCur, iq_w, iq_b, E, D);
      gemm(0, kbuf, varCur, ik_w, ik_b, E, D);
      gemm(0, vbuf, varCur, iv_w, iv_b, E, D);
      attn_kernel<128, 0, false><<<dim3(E / 16, 1, B), 32, 0, stream>>>(
          varNew, qbuf, kbuf, vbuf, nullptr, nullptr, E, E, scale);
      varCur = varNew;
    }
  }

  // ---- outputs: (obs, temp_he, var_he) concatenated f32 ----
  float* out = (float*)d_out;
  hipMemcpyAsync(out, obsCur, oBND * sizeof(float), hipMemcpyDeviceToDevice, stream);
  hipMemcpyAsync(out + oBND, tempCur, oBLD * sizeof(float), hipMemcpyDeviceToDevice, stream);
  hipMemcpyAsync(out + oBND + oBLD, varCur, oBED * sizeof(float), hipMemcpyDeviceToDevice,
                 stream);
}
